// Attention_Seqtovec_6133213299246
// MI455X (gfx1250) — matrix operand order repaired and checked by reading, not compiled, not run on hardware
//
#include <hip/hip_runtime.h>
#include <hip/hip_bf16.h>

// ---------------------------------------------------------------------------
// Problem constants (from the reference)
// ---------------------------------------------------------------------------
#define Bc   32
#define Tc   256
#define Sc   9
#define Ec   631
#define Dc   512
#define FFc  512
#define Hc   8
#define HDc  64
#define Lc   2
#define OUTc 512
#define NBT  (Bc * Tc)          // 8192
#define NROW (NBT * Sc)         // 73728 token rows
#define QKVN (3 * Dc)           // 1536

typedef __attribute__((ext_vector_type(16))) __bf16 v16bf;
typedef __attribute__((ext_vector_type(8)))  float  v8f;

// RNE fp32->bf16 (one-time weight conversion)
__device__ __forceinline__ unsigned short f2bf(float f) {
  union { float f; unsigned u; } c; c.f = f;
  unsigned u = c.u;
  u += 0x7fffu + ((u >> 16) & 1u);
  return (unsigned short)(u >> 16);
}

// fast pair pack fp32->2xbf16 (round half up) for the activation hot path
__device__ __forceinline__ unsigned pk2(float a, float b) {
  union { float f; unsigned u; } x, y; x.f = a; y.f = b;
  return ((x.u + 0x8000u) >> 16) | ((y.u + 0x8000u) & 0xffff0000u);
}

// ---------------------------------------------------------------------------
// Weight prep: fp32 K x N  ->  bf16 N x K (k-contiguous, WMMA-B friendly)
// ---------------------------------------------------------------------------
__global__ void cvt_t_kernel(const float* __restrict__ src,      // K x N
                             unsigned short* __restrict__ dst,   // N x K
                             int K, int N) {
  int i = blockIdx.x * blockDim.x + threadIdx.x;
  if (i < K * N) {
    int n = i / K, k = i - n * K;
    dst[i] = f2bf(src[(size_t)k * N + n]);
  }
}

// ---------------------------------------------------------------------------
// Embedding: h[row][:] = adapt_w[x[row]] + adapt_b   (row = bt*S + s)
// ---------------------------------------------------------------------------
__global__ __launch_bounds__(128) void embed_kernel(
    const int* __restrict__ x, const float* __restrict__ W,
    const float* __restrict__ bias, float* __restrict__ h) {
  const int row = blockIdx.x;
  const int t = threadIdx.x;
  const int xv = x[row];
  float4 w = ((const float4*)(W + (size_t)xv * Dc))[t];
  float4 b = ((const float4*)bias)[t];
  float4 o = make_float4(w.x + b.x, w.y + b.y, w.z + b.z, w.w + b.w);
  ((float4*)(h + (size_t)row * Dc))[t] = o;
}

// ---------------------------------------------------------------------------
// raw_emb: raw[bt][:] = sum_s adapt2_w[s*E + x[bt,s]][:] + adapt2_b
// ---------------------------------------------------------------------------
__global__ __launch_bounds__(128) void rawemb_kernel(
    const int* __restrict__ x, const float* __restrict__ W2,
    const float* __restrict__ b2, float* __restrict__ raw) {
  const int bt = blockIdx.x;
  const int t = threadIdx.x;
  float4 acc = ((const float4*)b2)[t];
  #pragma unroll
  for (int s = 0; s < Sc; s++) {
    int idx = x[bt * Sc + s];
    float4 w = ((const float4*)(W2 + ((size_t)s * Ec + idx) * OUTc))[t];
    acc.x += w.x; acc.y += w.y; acc.z += w.z; acc.w += w.w;
  }
  ((float4*)(raw + (size_t)bt * OUTc))[t] = acc;
}

// ---------------------------------------------------------------------------
// WMMA bf16 GEMM:  C[M,N] = A[M,K](f32, row stride lda) * W + bias
// W is bf16, pre-transposed N x K (k contiguous).  act: 0 none, 1 relu.
// Block tile 64x128x32, 256 threads = 8 waves, wave owns 32x32 (2x2 WMMA accs).
// Double-buffered LDS, one barrier per K-step.
// Requires M%64==0, N%128==0, K%32==0 (true for this model).
// ---------------------------------------------------------------------------
#define BM 64
#define BN 128
#define BK 32

__global__ __launch_bounds__(256) void gemm_bf16_kernel(
    const float* __restrict__ A, int lda,
    const unsigned short* __restrict__ B,   // bf16 bits, N x K row-major
    const float* __restrict__ bias,
    float* __restrict__ C, int ldc,
    int M, int N, int K, int act) {
  __shared__ unsigned short As[2][BM][BK];   // (m,k), k-pairs contiguous
  __shared__ unsigned short Bs[2][BN][BK];   // (n,k), k-pairs contiguous

  const int tid  = threadIdx.x;
  const int lane = tid & 31;
  const int wave = tid >> 5;        // 0..7
  const int wm   = wave >> 2;       // 0..1  (M direction)
  const int wn   = wave & 3;        // 0..3  (N direction)
  const int half = lane >> 4;       // hi/lo 16 lanes
  const int l15  = lane & 15;

  const int blockM = blockIdx.y * BM;
  const int blockN = blockIdx.x * BN;

  v8f acc[2][2] = {};

  // A staging: thread t: row = t/4, cols (t%4)*8..+7 (8 floats -> 8 bf16)
  const int arow = tid >> 2;
  const int acol = (tid & 3) * 8;
  // B staging: thread t: n = t/2, k-offset (t%2)*16 (16 bf16 contiguous)
  const int bn = tid >> 1;
  const int bk = (tid & 1) * 16;

  const float*          aptr = A + (size_t)(blockM + arow) * lda + acol;
  const unsigned short* bptr = B + (size_t)(blockN + bn) * K + bk;

  union FragU { v16bf v; unsigned u[8]; };

  // ---- prologue: stage tile 0
  {
    float4 f0 = *(const float4*)(aptr);
    float4 f1 = *(const float4*)(aptr + 4);
    uint4 aw;
    aw.x = pk2(f0.x, f0.y); aw.y = pk2(f0.z, f0.w);
    aw.z = pk2(f1.x, f1.y); aw.w = pk2(f1.z, f1.w);
    *(uint4*)&As[0][arow][acol] = aw;
    uint4 b0 = *(const uint4*)(bptr);
    uint4 b1 = *(const uint4*)(bptr + 8);
    *(uint4*)&Bs[0][bn][bk]       = b0;
    *(uint4*)(&Bs[0][bn][bk] + 8) = b1;
  }
  __syncthreads();

  const int nk = K / BK;
  for (int ks = 0; ks < nk; ks++) {
    const int cur = ks & 1;
    const int nxt = cur ^ 1;
    const bool have_next = (ks + 1 < nk);

    // ---- issue next tile's global loads early (latency hidden by WMMAs)
    float4 f0, f1; uint4 b0, b1;
    if (have_next) {
      const float* ap = aptr + (ks + 1) * BK;
      f0 = *(const float4*)(ap);
      f1 = *(const float4*)(ap + 4);
      const unsigned short* bp = bptr + (ks + 1) * BK;
      b0 = *(const uint4*)(bp);
      b1 = *(const uint4*)(bp + 8);
    }

    // ---- fragments from LDS[cur] (16-bit A 16x32 / B 32x16 lane layouts)
    FragU af[2];
    #pragma unroll
    for (int ma = 0; ma < 2; ma++) {
      const int row = wm * 32 + ma * 16 + l15;
      #pragma unroll
      for (int vv = 0; vv < 8; vv++) {
        const int kk = ((vv < 4) ? (2 * vv) : (2 * vv + 8)) + half * 8;
        af[ma].u[vv] = *(const unsigned*)&As[cur][row][kk];
      }
    }
    FragU bfm[2];
    #pragma unroll
    for (int nb = 0; nb < 2; nb++) {
      const int col = wn * 32 + nb * 16 + l15;
      const int kb = half * 8;
      #pragma unroll
      for (int vv = 0; vv < 8; vv++)
        bfm[nb].u[vv] = *(const unsigned*)&Bs[cur][col][kb + ((vv < 4) ? (2 * vv) : (2 * vv + 8))];
    }

    // ---- 4 WMMAs per K-step
    #pragma unroll
    for (int ma = 0; ma < 2; ma++)
      #pragma unroll
      for (int nb = 0; nb < 2; nb++)
        acc[ma][nb] = __builtin_amdgcn_wmma_f32_16x16x32_bf16(
            false, af[ma].v, false, bfm[nb].v, (short)0, acc[ma][nb],
            false, false);

    // ---- stage next tile into LDS[nxt]; prior reads of [nxt] were fenced
    // by the barrier at the end of the previous iteration.
    if (have_next) {
      uint4 aw;
      aw.x = pk2(f0.x, f0.y); aw.y = pk2(f0.z, f0.w);
      aw.z = pk2(f1.x, f1.y); aw.w = pk2(f1.z, f1.w);
      *(uint4*)&As[nxt][arow][acol] = aw;
      *(uint4*)&Bs[nxt][bn][bk]       = b0;
      *(uint4*)(&Bs[nxt][bn][bk] + 8) = b1;
    }
    __syncthreads();
  }

  // ---- epilogue: C/D layout: VGPR r -> M = r (+8 for hi lanes), N = lane
  #pragma unroll
  for (int ma = 0; ma < 2; ma++) {
    #pragma unroll
    for (int nb = 0; nb < 2; nb++) {
      const int gn = blockN + wn * 32 + nb * 16 + l15;
      const float bv = bias ? bias[gn] : 0.0f;
      #pragma unroll
      for (int r = 0; r < 8; r++) {
        const int gm = blockM + wm * 32 + ma * 16 + half * 8 + r;
        float v = acc[ma][nb][r] + bv;
        if (act == 1) v = fmaxf(v, 0.0f);
        C[(size_t)gm * ldc + gn] = v;
      }
    }
  }
}

// ---------------------------------------------------------------------------
// Attention: one block per sequence n (8 waves = 8 heads).
// qkv row layout: [q(512) | k(512) | v(512)], S=9 rows per sequence.
// ---------------------------------------------------------------------------
__global__ __launch_bounds__(256) void attn_kernel(
    const float* __restrict__ qkv, float* __restrict__ ctx) {
  __shared__ float sq[Hc][Sc][HDc];
  __shared__ float sk[Hc][Sc][HDc];
  __shared__ float sv[Hc][Sc][HDc];
  __shared__ float ss[Hc][Sc][12];

  const int n = blockIdx.x;
  const int t = threadIdx.x;
  const int hd = t >> 5;     // head = wave
  const int lane = t & 31;

  for (int idx = lane; idx < Sc * HDc; idx += 32) {
    const int s = idx >> 6, d = idx & 63;
    const size_t rb = ((size_t)(n * Sc + s)) * QKVN + hd * HDc + d;
    sq[hd][s][d] = qkv[rb];
    sk[hd][s][d] = qkv[rb + Dc];
    sv[hd][s][d] = qkv[rb + 2 * Dc];
  }
  __syncthreads();

  for (int p = lane; p < Sc * Sc; p += 32) {
    const int i = p / Sc, j = p - i * Sc;
    float dot = 0.f;
    #pragma unroll
    for (int d = 0; d < HDc; d++) dot += sq[hd][i][d] * sk[hd][j][d];
    ss[hd][i][j] = dot * 0.125f;   // 1/sqrt(64)
  }
  __syncthreads();

  if (lane < Sc) {
    float m = -1e30f;
    #pragma unroll
    for (int j = 0; j < Sc; j++) m = fmaxf(m, ss[hd][lane][j]);
    float e[Sc], sum = 0.f;
    #pragma unroll
    for (int j = 0; j < Sc; j++) { e[j] = __expf(ss[hd][lane][j] - m); sum += e[j]; }
    const float inv = 1.0f / sum;
    #pragma unroll
    for (int j = 0; j < Sc; j++) ss[hd][lane][j] = e[j] * inv;
  }
  __syncthreads();

  for (int idx = lane; idx < Sc * HDc; idx += 32) {
    const int s = idx >> 6, d = idx & 63;
    float o = 0.f;
    #pragma unroll
    for (int j = 0; j < Sc; j++) o += ss[hd][s][j] * sv[hd][j][d];
    ctx[((size_t)(n * Sc + s)) * Dc + hd * HDc + d] = o;
  }
}

// ---------------------------------------------------------------------------
// Fused residual + layernorm: out = LN(x + y) * g + b   (one block per row)
// ---------------------------------------------------------------------------
__global__ __launch_bounds__(256) void add_ln_kernel(
    const float* __restrict__ x, const float* __restrict__ y,
    const float* __restrict__ g, const float* __restrict__ b,
    float* __restrict__ out) {
  const int row = blockIdx.x;
  const int t = threadIdx.x;
  const size_t base = (size_t)row * Dc;

  float v0 = x[base + t]       + y[base + t];
  float v1 = x[base + t + 256] + y[base + t + 256];
  float s  = v0 + v1;
  float s2 = v0 * v0 + v1 * v1;
  #pragma unroll
  for (int o = 16; o > 0; o >>= 1) {
    s  += __shfl_down(s,  o, 32);
    s2 += __shfl_down(s2, o, 32);
  }
  __shared__ float red[16];
  const int wave = t >> 5, lane = t & 31;
  if (lane == 0) { red[wave] = s; red[8 + wave] = s2; }
  __syncthreads();
  if (t == 0) {
    float ts = 0.f, ts2 = 0.f;
    #pragma unroll
    for (int i = 0; i < 8; i++) { ts += red[i]; ts2 += red[8 + i]; }
    red[0] = ts; red[8] = ts2;
  }
  __syncthreads();
  const float mean = red[0] * (1.0f / Dc);
  const float var  = red[8] * (1.0f / Dc) - mean * mean;
  const float rstd = rsqrtf(var + 1e-5f);
  out[base + t]       = (v0 - mean) * rstd * g[t]       + b[t];
  out[base + t + 256] = (v1 - mean) * rstd * g[t + 256] + b[t + 256];
}

// ---------------------------------------------------------------------------
// Final: out = raw * (1 + relu(ov))
// ---------------------------------------------------------------------------
__global__ void final_kernel(const float* __restrict__ raw,
                             const float* __restrict__ ov,
                             float* __restrict__ out, int n) {
  int i = blockIdx.x * blockDim.x + threadIdx.x;
  if (i < n) out[i] = raw[i] * (1.0f + fmaxf(ov[i], 0.0f));
}

// ---------------------------------------------------------------------------
// Host orchestration
// ---------------------------------------------------------------------------
extern "C" void kernel_launch(void* const* d_in, const int* in_sizes, int n_in,
                              void* d_out, int out_size, void* d_ws, size_t ws_size,
                              hipStream_t stream) {
  (void)in_sizes; (void)n_in; (void)out_size; (void)ws_size;

  const int*   x        = (const int*)d_in[0];
  const float* adapt2_w = (const float*)d_in[1];
  const float* adapt2_b = (const float*)d_in[2];
  const float* adapt_w  = (const float*)d_in[3];
  const float* adapt_b  = (const float*)d_in[4];
  const float* qkv_w    = (const float*)d_in[5];
  const float* qkv_b    = (const float*)d_in[6];
  const float* out_w    = (const float*)d_in[7];
  const float* out_b    = (const float*)d_in[8];
  const float* ln1_g    = (const float*)d_in[9];
  const float* ln1_b    = (const float*)d_in[10];
  const float* lin1_w   = (const float*)d_in[11];
  const float* lin1_b   = (const float*)d_in[12];
  const float* lin2_w   = (const float*)d_in[13];
  const float* lin2_b   = (const float*)d_in[14];
  const float* ln2_g    = (const float*)d_in[15];
  const float* ln2_b    = (const float*)d_in[16];
  const float* fc_w     = (const float*)d_in[17];
  const float* fc_b     = (const float*)d_in[18];

  // workspace partition
  size_t off = 0;
  auto alloc = [&](size_t bytes) -> char* {
    char* p = (char*)d_ws + off;
    off += (bytes + 255) & ~(size_t)255;
    return p;
  };
  float* hbuf   = (float*)alloc((size_t)NROW * Dc * 4);       // current hidden
  float* qkvbuf = (float*)alloc((size_t)NROW * QKVN * 4);     // qkv, then 3 D-wide slices
  float* ctxbuf = (float*)alloc((size_t)NROW * Dc * 4);       // ctx / h_mid
  float* rawbuf = (float*)alloc((size_t)NBT * OUTc * 4);
  float* ovbuf  = (float*)alloc((size_t)NBT * OUTc * 4);
  unsigned short* wqkv  = (unsigned short*)alloc((size_t)Lc * Dc * QKVN * 2);
  unsigned short* wout  = (unsigned short*)alloc((size_t)Lc * Dc * Dc * 2);
  unsigned short* wlin1 = (unsigned short*)alloc((size_t)Lc * Dc * FFc * 2);
  unsigned short* wlin2 = (unsigned short*)alloc((size_t)Lc * FFc * Dc * 2);
  unsigned short* wfc   = (unsigned short*)alloc((size_t)Dc * OUTc * 2);

  float* slice0 = qkvbuf;                          // reused once qkv consumed
  float* slice1 = qkvbuf + (size_t)NROW * Dc;
  float* slice2 = qkvbuf + (size_t)2 * NROW * Dc;

  // fp32 (K x N) -> bf16 (N x K) weight prep
  auto cvt_t = [&](const float* src, unsigned short* dst, int K, int N) {
    int n = K * N;
    cvt_t_kernel<<<(n + 255) / 256, 256, 0, stream>>>(src, dst, K, N);
  };
  auto gemm = [&](const float* A, int lda, const unsigned short* Bw,
                  const float* bias, float* C, int ldc,
                  int M, int N, int K, int act) {
    dim3 grid(N / BN, M / BM);
    gemm_bf16_kernel<<<grid, 256, 0, stream>>>(A, lda, Bw, bias, C, ldc, M, N, K, act);
  };

  // ---- one-time weight conversion (bf16, transposed to N x K)
  for (int l = 0; l < Lc; l++) {
    cvt_t(qkv_w  + (size_t)l * Dc * QKVN, wqkv  + (size_t)l * Dc * QKVN, Dc, QKVN);
    cvt_t(out_w  + (size_t)l * Dc * Dc,   wout  + (size_t)l * Dc * Dc,   Dc, Dc);
    cvt_t(lin1_w + (size_t)l * Dc * FFc,  wlin1 + (size_t)l * Dc * FFc,  Dc, FFc);
    cvt_t(lin2_w + (size_t)l * FFc * Dc,  wlin2 + (size_t)l * FFc * Dc,  FFc, Dc);
  }
  cvt_t(fc_w, wfc, Dc, OUTc);

  // ---- embeddings
  embed_kernel<<<NROW, 128, 0, stream>>>(x, adapt_w, adapt_b, hbuf);
  rawemb_kernel<<<NBT, 128, 0, stream>>>(x, adapt2_w, adapt2_b, rawbuf);

  // ---- encoder layers
  for (int l = 0; l < Lc; l++) {
    gemm(hbuf, Dc, wqkv + (size_t)l * Dc * QKVN, qkv_b + l * QKVN,
         qkvbuf, QKVN, NROW, QKVN, Dc, 0);
    attn_kernel<<<NBT, 256, 0, stream>>>(qkvbuf, ctxbuf);
    gemm(ctxbuf, Dc, wout + (size_t)l * Dc * Dc, out_b + l * Dc,
         slice0, Dc, NROW, Dc, Dc, 0);
    add_ln_kernel<<<NROW, 256, 0, stream>>>(hbuf, slice0,
                                            ln1_g + l * Dc, ln1_b + l * Dc, ctxbuf);
    gemm(ctxbuf, Dc, wlin1 + (size_t)l * Dc * FFc, lin1_b + l * FFc,
         slice1, FFc, NROW, FFc, Dc, 1 /*relu*/);
    gemm(slice1, FFc, wlin2 + (size_t)l * FFc * Dc, lin2_b + l * Dc,
         slice2, Dc, NROW, Dc, FFc, 0);
    add_ln_kernel<<<NROW, 256, 0, stream>>>(ctxbuf, slice2,
                                            ln2_g + l * Dc, ln2_b + l * Dc, hbuf);
  }

  // ---- classifier on CLS rows (row stride S*D picks h[:,0,:] for free)
  gemm(hbuf, Sc * Dc, wfc, fc_b, ovbuf, OUTc, NBT, OUTc, Dc, 0);

  // ---- out = raw * (1 + relu(ov))
  final_kernel<<<(NBT * OUTc + 255) / 256, 256, 0, stream>>>(
      rawbuf, ovbuf, (float*)d_out, NBT * OUTc);
}
